// Voxurf_52243982189210
// MI455X (gfx1250) — matrix operand order repaired and checked by reading, not compiled, not run on hardware
//
#include <hip/hip_runtime.h>
#include <hip/hip_bf16.h>
#include <math.h>

#define GRIDN 160
#define NVOX  (GRIDN*GRIDN*GRIDN)        // 4,096,000
#define NPTS  (1<<20)                    // 1,048,576
#define NRAYS 8192
#define DIST  0.005f

typedef __attribute__((ext_vector_type(16))) _Float16 v16h;
typedef __attribute__((ext_vector_type(8)))  _Float16 v8h;
typedef __attribute__((ext_vector_type(4)))  _Float16 v4h;
typedef __attribute__((ext_vector_type(8)))  float    v8f;

__device__ __forceinline__ float vox_sigmoid(float x) { return 1.0f / (1.0f + __expf(-x)); }

// ---------------------------------------------------------------------------
// Kernel 1: central-difference gradient grid (matches jnp.gradient(vol)/h),
// stored as float4 per voxel so the main kernel gathers one b128 per corner.
// ---------------------------------------------------------------------------
__global__ void vox_gvol_kernel(const float* __restrict__ sdf, float4* __restrict__ g4) {
    int i = blockIdx.x * 256 + threadIdx.x;
    if (i >= NVOX) return;
    int z = i % GRIDN;
    int y = (i / GRIDN) % GRIDN;
    int x = i / (GRIDN * GRIDN);
#define VOXV(xx,yy,zz) sdf[((xx)*GRIDN+(yy))*GRIDN+(zz)]
    float gx = (x == 0)       ? (VOXV(1,y,z) - VOXV(0,y,z))
             : (x == GRIDN-1) ? (VOXV(GRIDN-1,y,z) - VOXV(GRIDN-2,y,z))
             : (VOXV(x+1,y,z) - VOXV(x-1,y,z)) * 0.5f;
    float gy = (y == 0)       ? (VOXV(x,1,z) - VOXV(x,0,z))
             : (y == GRIDN-1) ? (VOXV(x,GRIDN-1,z) - VOXV(x,GRIDN-2,z))
             : (VOXV(x,y+1,z) - VOXV(x,y-1,z)) * 0.5f;
    float gz = (z == 0)       ? (VOXV(x,y,1) - VOXV(x,y,0))
             : (z == GRIDN-1) ? (VOXV(x,y,GRIDN-1) - VOXV(x,y,GRIDN-2))
             : (VOXV(x,y,z+1) - VOXV(x,y,z-1)) * 0.5f;
#undef VOXV
    const float invh = (GRIDN - 1) * 0.5f;   // 1/h, h = 2/(GRID-1)
    g4[i] = make_float4(gx * invh, gy * invh, gz * invh, 0.0f);
}

// ---------------------------------------------------------------------------
// Kernel 2: k0 volume [12][X][Y][Z] f32 -> channel-last f16 [X][Y][Z][12]
// (24 B/voxel: 98 MB total -> fits L2; corner gather = 3x 8B vector loads)
// ---------------------------------------------------------------------------
__global__ void vox_k0pack_kernel(const float* __restrict__ k0, _Float16* __restrict__ k0h) {
    int e = blockIdx.x * 256 + threadIdx.x;
    if (e >= 12 * NVOX) return;
    int vox = e / 12;
    int c   = e - vox * 12;
    k0h[e] = (_Float16)k0[(size_t)c * NVOX + vox];
}

// ---------------------------------------------------------------------------
// Kernel 3: pack w0/w1/w2 into f16 WMMA B-matrix fragments (32x16, K x N).
// B layout (wave32, 16-bit): lane l holds column n = l&15; element j of lane
// half h holds K = 8h+j (j<8) and K = 16+8h+(j-8) (j>=8).
// Fragment linear layout: [frag][lane][16 halves] -> two b128 per lane.
// Layer0: K padded 75->96 (3 ktiles) x 8 ntiles = 24 frags
// Layer1: 4 x 8 = 32 frags ; Layer2: N padded 3->16, 4 x 1 = 4 frags
// ---------------------------------------------------------------------------
__global__ void vox_wpack_kernel(const float* __restrict__ w0,
                                 const float* __restrict__ w1,
                                 const float* __restrict__ w2,
                                 _Float16* __restrict__ wf) {
    int idx = blockIdx.x * 256 + threadIdx.x;      // 0 .. 60*512-1
    if (idx >= 60 * 512) return;
    int frag = idx >> 9;
    int r    = idx & 511;
    int lane = r >> 4;
    int j    = r & 15;
    const float* W; int Kdim, Ncols, NT, f;
    if (frag < 24)      { W = w0; Kdim = 75;  Ncols = 128; NT = 8; f = frag; }
    else if (frag < 56) { W = w1; Kdim = 128; Ncols = 128; NT = 8; f = frag - 24; }
    else                { W = w2; Kdim = 128; Ncols = 3;   NT = 1; f = frag - 56; }
    int kt = f / NT, nt = f - kt * NT;
    int n = nt * 16 + (lane & 15);
    int k = kt * 32 + (j >> 3) * 16 + ((lane >> 4) & 1) * 8 + (j & 7);
    float v = (k < Kdim && n < Ncols) ? W[(size_t)k * Ncols + n] : 0.0f;
    wf[idx] = (_Float16)v;
}

// ---------------------------------------------------------------------------
// Kernel 4: fused per-point kernel.  64 threads = 2 waves; each wave owns 32
// points.  Gathers + NeuS alpha per lane, features -> LDS, then a 3-layer MLP
// via v_wmma_f32_16x16x32_f16 (A fragments from LDS, B fragments from the
// pre-packed global buffer, f32 accumulation, ReLU/f16 between layers).
// ---------------------------------------------------------------------------
__global__ __launch_bounds__(64) void vox_main_kernel(
        const float* __restrict__ ray_pts, const float* __restrict__ rays_d,
        const float* __restrict__ sdf, const float4* __restrict__ g4,
        const _Float16* __restrict__ k0h, const _Float16* __restrict__ wf,
        const float* __restrict__ b0, const float* __restrict__ b1,
        const float* __restrict__ b2, const float* __restrict__ s_val,
        const int* __restrict__ ray_id,
        float* __restrict__ alpha_out, float* __restrict__ rgb_out) {

    const int wave = threadIdx.x >> 5;
    const int lane = threadIdx.x & 31;
    const int p    = blockIdx.x * 64 + threadIdx.x;

    __shared__ __align__(16) _Float16 sFeat[2][32 * 96];     // 12 KB
    __shared__ __align__(16) _Float16 sH[2][2][32 * 128];    // 32 KB

    // ---- per-lane point features -------------------------------------------
    float px = ray_pts[3 * p + 0], py = ray_pts[3 * p + 1], pz = ray_pts[3 * p + 2];
    int rid = ray_id[p];
    float ddx = rays_d[3 * rid + 0], ddy = rays_d[3 * rid + 1], ddz = rays_d[3 * rid + 2];

    float x01 = (px + 1.0f) * 0.5f, y01 = (py + 1.0f) * 0.5f, z01 = (pz + 1.0f) * 0.5f;
    float ux = fminf(fmaxf(x01 * 159.0f, 0.0f), 159.0f);
    float uy = fminf(fmaxf(y01 * 159.0f, 0.0f), 159.0f);
    float uz = fminf(fmaxf(z01 * 159.0f, 0.0f), 159.0f);
    int ix = (int)floorf(ux); ix = ix > 158 ? 158 : ix;
    int iy = (int)floorf(uy); iy = iy > 158 ? 158 : iy;
    int iz = (int)floorf(uz); iz = iz > 158 ? 158 : iz;
    float fx = ux - (float)ix, fy = uy - (float)iy, fz = uz - (float)iz;

    float sdfv = 0.0f, gx = 0.0f, gy = 0.0f, gz = 0.0f;
    float k0a[12];
#pragma unroll
    for (int j = 0; j < 12; ++j) k0a[j] = 0.0f;

#pragma unroll
    for (int c = 0; c < 8; ++c) {
        int cx = (c >> 2) & 1, cy = (c >> 1) & 1, cz = c & 1;
        float w = (cx ? fx : 1.0f - fx) * (cy ? fy : 1.0f - fy) * (cz ? fz : 1.0f - fz);
        int vi = ((ix + cx) * GRIDN + (iy + cy)) * GRIDN + (iz + cz);
        sdfv += w * sdf[vi];
        float4 g = g4[vi];
        gx += w * g.x; gy += w * g.y; gz += w * g.z;
        const v4h* kp = (const v4h*)&k0h[(size_t)vi * 12];
        v4h a0 = kp[0], a1 = kp[1], a2 = kp[2];
#pragma unroll
        for (int j = 0; j < 4; ++j) {
            k0a[j]     += w * (float)a0[j];
            k0a[4 + j] += w * (float)a1[j];
            k0a[8 + j] += w * (float)a2[j];
        }
    }

    float gl = sqrtf(gx * gx + gy * gy + gz * gz) + 1e-8f;
    float nx = gx / gl, ny = gy / gl, nz = gz / gl;
    float true_cos = ddx * nx + ddy * ny + ddz * nz;
    float iter_cos = fminf(true_cos, 0.0f);
    float inv_s = 1.0f / s_val[0];
    float pc = vox_sigmoid((sdfv - iter_cos * DIST * 0.5f) * inv_s);
    float nc = vox_sigmoid((sdfv + iter_cos * DIST * 0.5f) * inv_s);
    float alpha = fminf(fmaxf((pc - nc + 1e-5f) / (pc + 1e-5f), 0.0f), 1.0f);
    alpha_out[p] = alpha;

    // ---- features -> LDS row (75 used, pad to 96) --------------------------
    _Float16* Fr = &sFeat[wave][lane * 96];
#pragma unroll
    for (int j = 0; j < 12; ++j) Fr[j] = (_Float16)k0a[j];
    Fr[12] = (_Float16)x01; Fr[13] = (_Float16)y01; Fr[14] = (_Float16)z01;
    {
        float xs[3] = {x01, y01, z01};
#pragma unroll
        for (int a = 0; a < 3; ++a)
#pragma unroll
            for (int f = 0; f < 5; ++f) {
                float s, c; __sincosf(xs[a] * (float)(1 << f), &s, &c);
                Fr[15 + a * 5 + f] = (_Float16)s;
                Fr[30 + a * 5 + f] = (_Float16)c;
            }
    }
    Fr[45] = (_Float16)ddx; Fr[46] = (_Float16)ddy; Fr[47] = (_Float16)ddz;
    {
        float vs[3] = {ddx, ddy, ddz};
#pragma unroll
        for (int a = 0; a < 3; ++a)
#pragma unroll
            for (int f = 0; f < 4; ++f) {
                float s, c; __sincosf(vs[a] * (float)(1 << f), &s, &c);
                Fr[48 + a * 4 + f] = (_Float16)s;
                Fr[60 + a * 4 + f] = (_Float16)c;
            }
    }
    Fr[72] = (_Float16)nx; Fr[73] = (_Float16)ny; Fr[74] = (_Float16)nz;
#pragma unroll
    for (int j = 75; j < 96; ++j) Fr[j] = (_Float16)0.0f;
    __syncthreads();

    // ---- WMMA MLP ----------------------------------------------------------
    const int klo   = ((lane >> 4) & 1) * 8;   // A fragment K sub-offset
    const int nloc  = lane & 15;
    const int moff  = ((lane >> 4) & 1) * 8;   // D fragment M sub-offset
    const int pbase = blockIdx.x * 64 + wave * 32;
    _Float16* F  = sFeat[wave];
    _Float16* H0 = sH[wave][0];
    _Float16* H1 = sH[wave][1];

    // Layer 0: [32x96] x [96x128]
#pragma unroll
    for (int mt = 0; mt < 2; ++mt) {
        int row = mt * 16 + nloc;  // A-matrix row = lane&15 within M-tile
#pragma unroll
        for (int nt = 0; nt < 8; ++nt) {
            float bn = b0[nt * 16 + nloc];
            v8f acc;
#pragma unroll
            for (int r = 0; r < 8; ++r) acc[r] = bn;
#pragma unroll
            for (int kt = 0; kt < 3; ++kt) {
                v16h a, b;
                ((v8h*)&a)[0] = *(const v8h*)&F[row * 96 + kt * 32 + klo];
                ((v8h*)&a)[1] = *(const v8h*)&F[row * 96 + kt * 32 + 16 + klo];
                const _Float16* fb = &wf[(size_t)(kt * 8 + nt) * 512 + lane * 16];
                ((v8h*)&b)[0] = *(const v8h*)fb;
                ((v8h*)&b)[1] = *(const v8h*)(fb + 8);
                acc = __builtin_amdgcn_wmma_f32_16x16x32_f16(false, a, false, b,
                                                             (short)0, acc, false, false);
            }
            int mbase = mt * 16 + moff;
#pragma unroll
            for (int r = 0; r < 8; ++r) {
                float v = acc[r]; v = v > 0.0f ? v : 0.0f;
                H0[(mbase + r) * 128 + nt * 16 + nloc] = (_Float16)v;
            }
        }
    }
    __syncthreads();

    // Layer 1: [32x128] x [128x128]
#pragma unroll
    for (int mt = 0; mt < 2; ++mt) {
        int row = mt * 16 + nloc;
#pragma unroll
        for (int nt = 0; nt < 8; ++nt) {
            float bn = b1[nt * 16 + nloc];
            v8f acc;
#pragma unroll
            for (int r = 0; r < 8; ++r) acc[r] = bn;
#pragma unroll
            for (int kt = 0; kt < 4; ++kt) {
                v16h a, b;
                ((v8h*)&a)[0] = *(const v8h*)&H0[row * 128 + kt * 32 + klo];
                ((v8h*)&a)[1] = *(const v8h*)&H0[row * 128 + kt * 32 + 16 + klo];
                const _Float16* fb = &wf[(size_t)(24 + kt * 8 + nt) * 512 + lane * 16];
                ((v8h*)&b)[0] = *(const v8h*)fb;
                ((v8h*)&b)[1] = *(const v8h*)(fb + 8);
                acc = __builtin_amdgcn_wmma_f32_16x16x32_f16(false, a, false, b,
                                                             (short)0, acc, false, false);
            }
            int mbase = mt * 16 + moff;
#pragma unroll
            for (int r = 0; r < 8; ++r) {
                float v = acc[r]; v = v > 0.0f ? v : 0.0f;
                H1[(mbase + r) * 128 + nt * 16 + nloc] = (_Float16)v;
            }
        }
    }
    __syncthreads();

    // Layer 2: [32x128] x [128x16] (cols 0..2 valid), sigmoid, store rgb
    {
        float bn = (nloc < 3) ? b2[nloc] : 0.0f;
#pragma unroll
        for (int mt = 0; mt < 2; ++mt) {
            int row = mt * 16 + nloc;
            v8f acc;
#pragma unroll
            for (int r = 0; r < 8; ++r) acc[r] = bn;
#pragma unroll
            for (int kt = 0; kt < 4; ++kt) {
                v16h a, b;
                ((v8h*)&a)[0] = *(const v8h*)&H1[row * 128 + kt * 32 + klo];
                ((v8h*)&a)[1] = *(const v8h*)&H1[row * 128 + kt * 32 + 16 + klo];
                const _Float16* fb = &wf[(size_t)(56 + kt) * 512 + lane * 16];
                ((v8h*)&b)[0] = *(const v8h*)fb;
                ((v8h*)&b)[1] = *(const v8h*)(fb + 8);
                acc = __builtin_amdgcn_wmma_f32_16x16x32_f16(false, a, false, b,
                                                             (short)0, acc, false, false);
            }
            if (nloc < 3) {
                int mbase = mt * 16 + moff;
#pragma unroll
                for (int r = 0; r < 8; ++r) {
                    float v = 1.0f / (1.0f + expf(-acc[r]));
                    rgb_out[(size_t)(pbase + mbase + r) * 3 + nloc] = v;
                }
            }
        }
    }
}

// ---------------------------------------------------------------------------
// Kernel 5: ragged per-ray compositing (exclusive cumprod in log space),
// one thread per ray, sample range found by binary search on sorted ray_id.
// ---------------------------------------------------------------------------
__device__ __forceinline__ int vox_lower_bound(const int* a, int n, int v) {
    int lo = 0, hi = n;
    while (lo < hi) { int m = (lo + hi) >> 1; if (a[m] < v) lo = m + 1; else hi = m; }
    return lo;
}

__global__ void vox_composite_kernel(const int* __restrict__ ray_id,
                                     const float* __restrict__ alpha,
                                     const float* __restrict__ rgb,
                                     float* __restrict__ out) {
    int r = blockIdx.x * 256 + threadIdx.x;
    if (r >= NRAYS) return;
    int start = vox_lower_bound(ray_id, NPTS, r);
    int end   = vox_lower_bound(ray_id, NPTS, r + 1);
    float logT = 0.0f, o0 = 0.0f, o1 = 0.0f, o2 = 0.0f;
    for (int i = start; i < end; ++i) {
        float a = alpha[i];
        float T = expf(logT);
        float w = a * T;
        o0 += w * rgb[3 * i + 0];
        o1 += w * rgb[3 * i + 1];
        o2 += w * rgb[3 * i + 2];
        logT += log1pf(-fminf(a, 1.0f - 1e-7f));
    }
    float ail = expf(logT);
    out[3 * r + 0] = o0 + ail;
    out[3 * r + 1] = o1 + ail;
    out[3 * r + 2] = o2 + ail;
}

// ---------------------------------------------------------------------------
extern "C" void kernel_launch(void* const* d_in, const int* in_sizes, int n_in,
                              void* d_out, int out_size, void* d_ws, size_t ws_size,
                              hipStream_t stream) {
    const float* ray_pts = (const float*)d_in[0];
    const float* rays_d  = (const float*)d_in[1];
    const float* sdf     = (const float*)d_in[2];
    const float* k0      = (const float*)d_in[3];
    const float* w0      = (const float*)d_in[4];
    const float* b0      = (const float*)d_in[5];
    const float* w1      = (const float*)d_in[6];
    const float* b1      = (const float*)d_in[7];
    const float* w2      = (const float*)d_in[8];
    const float* b2      = (const float*)d_in[9];
    const float* s_val   = (const float*)d_in[10];
    const int*   ray_id  = (const int*)d_in[11];
    float* out = (float*)d_out;

    char* ws = (char*)d_ws;
    float4*    g4    = (float4*)(ws + 0);                    //  65,536,000 B
    _Float16*  k0h   = (_Float16*)(ws + 65536000ull);        //  98,304,000 B
    float*     alpha = (float*)(ws + 163840000ull);          //   4,194,304 B
    float*     rgb   = (float*)(ws + 168034304ull);          //  12,582,912 B
    _Float16*  wf    = (_Float16*)(ws + 180617216ull);       //      61,440 B

    vox_gvol_kernel<<<dim3((NVOX + 255) / 256), dim3(256), 0, stream>>>(sdf, g4);
    vox_k0pack_kernel<<<dim3((12 * NVOX + 255) / 256), dim3(256), 0, stream>>>(k0, k0h);
    vox_wpack_kernel<<<dim3(120), dim3(256), 0, stream>>>(w0, w1, w2, wf);
    vox_main_kernel<<<dim3(NPTS / 64), dim3(64), 0, stream>>>(
        ray_pts, rays_d, sdf, g4, k0h, wf, b0, b1, b2, s_val, ray_id, alpha, rgb);
    vox_composite_kernel<<<dim3(NRAYS / 256), dim3(256), 0, stream>>>(ray_id, alpha, rgb, out);
}
